// MultiHeadAttentionPooling1D_20684562497812
// MI455X (gfx1250) — compile-verified, matrix operand order checked by reading, not run on hardware
//
#include <hip/hip_runtime.h>
#include <hip/hip_bf16.h>
#include <math.h>

// Problem dims (fixed by the reference)
#define B  32
#define S  4096
#define D  128
#define H  4
#define ROWS (B * S)          // 131072
#define NTILES (ROWS / 16)    // 8192

typedef __bf16 bf16_t;
typedef bf16_t v16bf __attribute__((ext_vector_type(16)));
typedef float  v8f   __attribute__((ext_vector_type(8)));

// ---------------------------------------------------------------------------
// Kernel 0: convert Wk (f32, [k][n] row-major) -> bf16, transposed to [n][k]
// so each WMMA B-fragment lane reads two contiguous 16B chunks.
// ---------------------------------------------------------------------------
__global__ void cvt_wk_kernel(const float* __restrict__ Wk, bf16_t* __restrict__ WkT) {
    int i = blockIdx.x * blockDim.x + threadIdx.x;   // 0..16383
    int k = i >> 7;
    int n = i & 127;
    WkT[n * D + k] = (bf16_t)Wk[k * D + n];
}

// ---------------------------------------------------------------------------
// Kernel 1: logits = tanh(x @ Wk) @ Wo, masked.  One wave per 16-row tile.
//   A (16x32 bf16):  lanes 0-15 row M=lane, K = {k0..k0+7, k0+16..k0+23} (k0=32*kk)
//                    lanes 16-31 same rows, K halves shifted by 8.
//   B (32x16 bf16):  lane holds column N = lane&15, same K mapping (from WkT).
//   C (16x16 f32):   VGPR v, lanes 0-15 -> M=v, lanes 16-31 -> M=v+8, N=lane&15.
// ---------------------------------------------------------------------------
__global__ void __launch_bounds__(128)
logits_kernel(const float* __restrict__ x, const int* __restrict__ mask,
              const bf16_t* __restrict__ WkT, const float* __restrict__ Wo,
              float* __restrict__ wlog) {
    const int lane = threadIdx.x & 31;
    const int wid  = threadIdx.x >> 5;
    const int tile = blockIdx.x * (blockDim.x >> 5) + wid;
    const long row_base = (long)tile * 16;

    const int half = lane >> 4;   // 0 or 1: which K-half this lane carries
    const int l15  = lane & 15;

    v8f acc[8];
#pragma unroll
    for (int nn = 0; nn < 8; ++nn)
#pragma unroll
        for (int v = 0; v < 8; ++v) acc[nn][v] = 0.0f;

    const float* xrow = x + (row_base + l15) * D;   // this lane's A row

#pragma unroll
    for (int kk = 0; kk < 4; ++kk) {
        const int k0 = kk * 32 + half * 8;

        // ---- A fragment: 16 bf16 per lane, two contiguous 8-float runs ----
        v16bf a;
#pragma unroll
        for (int e = 0; e < 8; ++e) a[e]     = (bf16_t)xrow[k0 + e];
#pragma unroll
        for (int e = 0; e < 8; ++e) a[8 + e] = (bf16_t)xrow[k0 + 16 + e];

        // ---- sweep the 8 N-tiles of the 16x128 output ----
#pragma unroll
        for (int nn = 0; nn < 8; ++nn) {
            const bf16_t* wt = WkT + (nn * 16 + l15) * D;  // column n as a k-row
            v16bf b;
#pragma unroll
            for (int e = 0; e < 8; ++e) b[e]     = wt[k0 + e];
#pragma unroll
            for (int e = 0; e < 8; ++e) b[8 + e] = wt[k0 + 16 + e];

            acc[nn] = __builtin_amdgcn_wmma_f32_16x16x32_bf16(
                false, a, false, b, (short)0, acc[nn], false, false);
        }
    }

    // ---- tanh on the full 16x128 tile held in C-registers ----
#pragma unroll
    for (int nn = 0; nn < 8; ++nn)
#pragma unroll
        for (int v = 0; v < 8; ++v) acc[nn][v] = tanhf(acc[nn][v]);

    // ---- skinny GEMM: logits[m,h] = sum_n T[m,n] * Wo[n,h] ----
    // This lane owns n = nn*16 + l15; preload its 8 Wo rows (4 floats each).
    float wo0[8], wo1[8], wo2[8], wo3[8];
#pragma unroll
    for (int nn = 0; nn < 8; ++nn) {
        const float4 t = *(const float4*)(Wo + (nn * 16 + l15) * H);
        wo0[nn] = t.x; wo1[nn] = t.y; wo2[nn] = t.z; wo3[nn] = t.w;
    }

#pragma unroll
    for (int v = 0; v < 8; ++v) {
        float p0 = 0.f, p1 = 0.f, p2 = 0.f, p3 = 0.f;
#pragma unroll
        for (int nn = 0; nn < 8; ++nn) {
            const float t = acc[nn][v];
            p0 = fmaf(t, wo0[nn], p0);
            p1 = fmaf(t, wo1[nn], p1);
            p2 = fmaf(t, wo2[nn], p2);
            p3 = fmaf(t, wo3[nn], p3);
        }
        // butterfly reduction over the 16 lanes of each half (same row M)
#pragma unroll
        for (int off = 1; off < 16; off <<= 1) {
            p0 += __shfl_xor(p0, off, 32);
            p1 += __shfl_xor(p1, off, 32);
            p2 += __shfl_xor(p2, off, 32);
            p3 += __shfl_xor(p3, off, 32);
        }
        const long grow = row_base + v + 8 * half;   // global (b*S+s) row
        if (l15 == 0) {
            const float mk = (mask[grow] == 0) ? 1.0e12f : 0.0f;
            float4 out;
            out.x = p0 - mk; out.y = p1 - mk; out.z = p2 - mk; out.w = p3 - mk;
            *(float4*)(wlog + grow * H) = out;
        }
    }
}

// ---------------------------------------------------------------------------
// Kernel 2: softmax over S for each (b,h), in place on wlog [B*S, H].
// ---------------------------------------------------------------------------
__global__ void __launch_bounds__(256)
softmax_kernel(float* __restrict__ wlog) {
    const int b = blockIdx.x >> 2;
    const int h = blockIdx.x & 3;
    float* base = wlog + (size_t)b * S * H + h;   // stride H between s values
    __shared__ float red[256];
    const int t = threadIdx.x;

    // pass 1: max
    float m = -INFINITY;
    for (int s = t; s < S; s += 256) m = fmaxf(m, base[s * H]);
    red[t] = m; __syncthreads();
    for (int o = 128; o > 0; o >>= 1) {
        if (t < o) red[t] = fmaxf(red[t], red[t + o]);
        __syncthreads();
    }
    m = red[0]; __syncthreads();

    // pass 2: sum of exp
    float sum = 0.f;
    for (int s = t; s < S; s += 256) sum += __expf(base[s * H] - m);
    red[t] = sum; __syncthreads();
    for (int o = 128; o > 0; o >>= 1) {
        if (t < o) red[t] += red[t + o];
        __syncthreads();
    }
    sum = red[0];

    // pass 3: normalize
    const float inv = 1.0f / sum;
    for (int s = t; s < S; s += 256) base[s * H] = __expf(base[s * H] - m) * inv;
}

// ---------------------------------------------------------------------------
// Kernel 3: pooled[b,h,d] = sum_s w[b,s,h] * x[b,s,d].  One block per batch,
// 128 threads (thread = d), w chunks staged through LDS.
// ---------------------------------------------------------------------------
__global__ void __launch_bounds__(128)
pool_kernel(const float* __restrict__ x, const float* __restrict__ w,
            float* __restrict__ pooled) {
    const int b = blockIdx.x;
    const int d = threadIdx.x;             // 0..127
    __shared__ float wch[128 * H];

    const float* xb = x + (size_t)b * S * D;
    const float* wb = w + (size_t)b * S * H;

    float a0 = 0.f, a1 = 0.f, a2 = 0.f, a3 = 0.f;
    for (int s0 = 0; s0 < S; s0 += 128) {
        ((float4*)wch)[d] = *(const float4*)(wb + (size_t)(s0 + d) * H);
        __syncthreads();
#pragma unroll 4
        for (int si = 0; si < 128; ++si) {
            const float xv = xb[(size_t)(s0 + si) * D + d];
            a0 = fmaf(wch[si * H + 0], xv, a0);
            a1 = fmaf(wch[si * H + 1], xv, a1);
            a2 = fmaf(wch[si * H + 2], xv, a2);
            a3 = fmaf(wch[si * H + 3], xv, a3);
        }
        __syncthreads();
    }
    float* pb = pooled + (size_t)b * H * D;
    pb[0 * D + d] = a0;
    pb[1 * D + d] = a1;
    pb[2 * D + d] = a2;
    pb[3 * D + d] = a3;
}

// ---------------------------------------------------------------------------
extern "C" void kernel_launch(void* const* d_in, const int* in_sizes, int n_in,
                              void* d_out, int out_size, void* d_ws, size_t ws_size,
                              hipStream_t stream) {
    const float* x    = (const float*)d_in[0];   // [B,S,D] f32
    const int*   mask = (const int*)  d_in[1];   // [B,S]   i32
    const float* Wk   = (const float*)d_in[2];   // [D,D]   f32
    const float* Wo   = (const float*)d_in[3];   // [D,H]   f32

    float* pooled = (float*)d_out;               // [B, H*D] = 16384 floats
    float* wout   = pooled + B * H * D;          // [B,S,H]  = 524288 floats

    bf16_t* WkT = (bf16_t*)d_ws;                 // 128*128*2 = 32 KB scratch

    // 0) Wk -> bf16 transposed
    cvt_wk_kernel<<<(D * D) / 256, 256, 0, stream>>>(Wk, WkT);

    // 1) masked logits via bf16 WMMA (8192 tiles, 4 waves/block)
    logits_kernel<<<NTILES / 4, 128, 0, stream>>>(x, mask, WkT, Wo, wout);

    // 2) softmax over S per (b,h), in place
    softmax_kernel<<<B * H, 256, 0, stream>>>(wout);

    // 3) weighted-sum pooling
    pool_kernel<<<B, 128, 0, stream>>>(x, wout, pooled);
}
